// TFA_88089779241014
// MI455X (gfx1250) — compile-verified, matrix operand order checked by reading, not run on hardware
//
#include <hip/hip_runtime.h>

// Triangle attention: B=1, N=256, H=4, S=256, D=32 (all f32).
// Block = 128 threads (4 waves) = one (n,h) pair. Phase 1: cooperatively
// convert K and V to bf16 WMMA-B fragments in LDS (v_cvt_pk_bf16_f32).
// Phase 2: each wave runs 4 query tiles. Bias is pre-loaded into the WMMA
// C operand (scaled by sqrt(D)); softmax runs in log2 domain; fragment
// loads are software-pipelined past the v_wmma_f32_16x16x32_bf16 ops.

typedef __attribute__((ext_vector_type(16))) __bf16   v16bf;
typedef __attribute__((ext_vector_type(8)))  float    v8f;
typedef __attribute__((ext_vector_type(8)))  unsigned v8u;
typedef __attribute__((ext_vector_type(4)))  unsigned v4u;
typedef __attribute__((ext_vector_type(2)))  unsigned v2u;

#define NN    256
#define HEADS 4
#define SLEN  256
#define DDIM  32
#define LDP   264   // P row stride in bf16: dword stride 132 (mod 64 = 4)

static __device__ __forceinline__ unsigned pk2(float a, float b) {
  unsigned short x = __builtin_bit_cast(unsigned short, (__bf16)a);
  unsigned short y = __builtin_bit_cast(unsigned short, (__bf16)b);
  return (unsigned)x | ((unsigned)y << 16);
}

// K base for fragment dword-pair ip (i = 2*ip), hi = lane>>4
static __device__ __forceinline__ int kb4(int ip, int hi) {
  return (ip < 2 ? 4 * ip : 16 + 4 * (ip - 2)) + 8 * hi;
}

// read one 16x32 bf16 B fragment from LDS ([tile][ip][lane][2] dword layout)
static __device__ __forceinline__ v8u ldfrag(const unsigned* F, int tile, int lane) {
  v8u r;
  #pragma unroll
  for (int ip = 0; ip < 4; ++ip) {
    const v2u d2 = *(const v2u*)&F[((tile * 4 + ip) * 32 + lane) * 2];
    r[2 * ip]     = d2[0];
    r[2 * ip + 1] = d2[1];
  }
  return r;
}

__global__ __launch_bounds__(128)
void TFA_triattn_kernel(const float* __restrict__ q, const float* __restrict__ k,
                        const float* __restrict__ v, const float* __restrict__ mb,
                        const float* __restrict__ tb, float* __restrict__ out) {
  __shared__ unsigned KF[16 * 4 * 32 * 2];   // 16 KB  (key tiles)
  __shared__ unsigned VF[16 * 4 * 32 * 2];   // 16 KB  (tile = kc*2+nt)
  __shared__ __bf16   PS[4][16 * LDP];       // 33 KB  per-wave P staging

  const int tid  = threadIdx.x;
  const int lane = tid & 31;
  const int w    = tid >> 5;
  const int lo   = lane & 15;
  const int hi   = lane >> 4;
  const int h    = blockIdx.x;
  const int n    = blockIdx.y;

  const int base = (n * HEADS + h) * SLEN * DDIM;
  const float* Q  = q + base;
  const float* K  = k + base;
  const float* V  = v + base;
  const float* MB = mb + n * SLEN;
  float*       O0 = out + base;

  // ---------------- phase 1: K,V -> bf16 fragments in LDS ----------------
  {
    const int ipc = (tid >> 5) & 3;          // per-thread constants
    const int Lc  = tid & 31;
    const int llc = Lc & 15, hhc = Lc >> 4;
    const int kbc = kb4(ipc, hhc);
    #pragma unroll
    for (int t = 0; t < 16; ++t) {           // key tile t
      const float* src = K + (t * 16 + llc) * DDIM + kbc;   // 4 consecutive f32
      const int o = ((t * 4 + ipc) * 32 + Lc) * 2;
      KF[o]     = pk2(src[0], src[1]);
      KF[o + 1] = pk2(src[2], src[3]);
    }
    #pragma unroll
    for (int t = 0; t < 16; ++t) {           // t = kc*2 + nt
      const int kk = (t >> 1) * 32 + kbc;
      const int d  = (t & 1) * 16 + llc;
      const float* src = V + kk * DDIM + d;  // 4 rows, stride DDIM
      const int o = ((t * 4 + ipc) * 32 + Lc) * 2;
      VF[o]     = pk2(src[0],        src[DDIM]);
      VF[o + 1] = pk2(src[2 * DDIM], src[3 * DDIM]);
    }
  }
  __syncthreads();

  const float invsc = 5.656854249492381f;                       // sqrt(32)
  const float sigma = 0.17677669529663687f * 1.4426950408889634f; // scale*log2(e)

  // mask bias row (q-tile invariant), pre-scaled by sqrt(D)
  float mb2[16];
  #pragma unroll
  for (int t = 0; t < 16; ++t) mb2[t] = MB[t * 16 + lo] * invsc;

  __bf16* Pw = &PS[w][0];

  // ---------------- phase 2: 4 query tiles per wave ----------------
  #pragma unroll 1
  for (int qi = 0; qi < 4; ++qi) {
    const int qt = w * 4 + qi;
    const float* TB = tb + (h * SLEN + qt * 16) * SLEN;

    // Q tile -> bf16 A fragment
    v8u au;
    const int mrow = qt * 16 + lo;
    #pragma unroll
    for (int ip = 0; ip < 4; ++ip) {
      const float* src = Q + mrow * DDIM + kb4(ip, hi);
      au[2 * ip]     = pk2(src[0], src[1]);
      au[2 * ip + 1] = pk2(src[2], src[3]);
    }
    const v16bf aq = __builtin_bit_cast(v16bf, au);

    // scores: c[t] = Q*K^T + (mask+tri_bias)*sqrt(D), pipelined B-frag loads
    v8f c[16];
    v8u bu = ldfrag(KF, 0, lane);
    #pragma unroll
    for (int t = 0; t < 16; ++t) {
      const v8u bu_next = (t < 15) ? ldfrag(KF, t + 1, lane) : bu;
      // bias into WMMA C operand
      v8f ci;
      const int col = t * 16 + lo;
      #pragma unroll
      for (int r = 0; r < 8; ++r)
        ci[r] = fmaf(TB[(r + 8 * hi) * SLEN + col], invsc, mb2[t]);
      c[t] = __builtin_amdgcn_wmma_f32_16x16x32_bf16(
                 false, aq, false, __builtin_bit_cast(v16bf, bu),
                 (short)0, ci, false, false);
      bu = bu_next;
    }

    // row max over raw c (monotonic in final score)
    float rmax[8], rsum[8];
    #pragma unroll
    for (int r = 0; r < 8; ++r) rmax[r] = -3.0e38f;
    #pragma unroll
    for (int t = 0; t < 16; ++t)
      #pragma unroll
      for (int r = 0; r < 8; ++r) rmax[r] = fmaxf(rmax[r], c[t][r]);
    #pragma unroll
    for (int r = 0; r < 8; ++r) {
      #pragma unroll
      for (int m = 1; m < 16; m <<= 1)
        rmax[r] = fmaxf(rmax[r], __shfl_xor(rmax[r], m, 32));
      rsum[r] = 0.0f;
    }

    // p = exp2(c*sigma - rmax*sigma); sum rows; stage bf16 P in LDS
    float nm2[8];
    #pragma unroll
    for (int r = 0; r < 8; ++r) nm2[r] = -rmax[r] * sigma;
    #pragma unroll
    for (int t = 0; t < 16; ++t) {
      #pragma unroll
      for (int r = 0; r < 8; ++r) {
        const float p = __builtin_amdgcn_exp2f(fmaf(c[t][r], sigma, nm2[r]));
        rsum[r] += p;
        Pw[(r + 8 * hi) * LDP + t * 16 + lo] = (__bf16)p;
      }
    }
    #pragma unroll
    for (int r = 0; r < 8; ++r) {
      #pragma unroll
      for (int m = 1; m < 16; m <<= 1)
        rsum[r] += __shfl_xor(rsum[r], m, 32);
    }

    // out = P * V : K=256 in 8 chunks; pipelined ap (b128) + bv (b64) loads
    v8f acc0 = {}, acc1 = {};
    const v4u* P4 = (const v4u*)Pw;
    v4u a0 = P4[(lo * (LDP / 2) + 4 * hi) >> 2];
    v4u a1 = P4[((lo * (LDP / 2) + 4 * hi) >> 2) + 2];
    #pragma unroll
    for (int kc = 0; kc < 8; ++kc) {
      const int nb = (lo * (LDP / 2) + (kc + 1) * 16 + 4 * hi) >> 2;
      const v4u a0n = (kc < 7) ? P4[nb]     : a0;
      const v4u a1n = (kc < 7) ? P4[nb + 2] : a1;
      const v8u bv0 = ldfrag(VF, kc * 2,     lane);
      const v8u bv1 = ldfrag(VF, kc * 2 + 1, lane);
      v8u apu;
      apu[0] = a0[0]; apu[1] = a0[1]; apu[2] = a0[2]; apu[3] = a0[3];
      apu[4] = a1[0]; apu[5] = a1[1]; apu[6] = a1[2]; apu[7] = a1[3];
      const v16bf ap = __builtin_bit_cast(v16bf, apu);
      acc0 = __builtin_amdgcn_wmma_f32_16x16x32_bf16(
                 false, ap, false, __builtin_bit_cast(v16bf, bv0),
                 (short)0, acc0, false, false);
      acc1 = __builtin_amdgcn_wmma_f32_16x16x32_bf16(
                 false, ap, false, __builtin_bit_cast(v16bf, bv1),
                 (short)0, acc1, false, false);
      a0 = a0n; a1 = a1n;
    }

    // normalize (fast rcp) and store f32
    float* O = O0 + qt * 16 * DDIM;
    #pragma unroll
    for (int r = 0; r < 8; ++r) {
      const int row  = r + 8 * hi;
      const float ri = __builtin_amdgcn_rcpf(rsum[r]);
      O[row * DDIM + lo]      = acc0[r] * ri;
      O[row * DDIM + 16 + lo] = acc1[r] * ri;
    }
  }
}

extern "C" void kernel_launch(void* const* d_in, const int* in_sizes, int n_in,
                              void* d_out, int out_size, void* d_ws, size_t ws_size,
                              hipStream_t stream) {
  (void)in_sizes; (void)n_in; (void)out_size; (void)d_ws; (void)ws_size;
  const float* q   = (const float*)d_in[0];
  const float* k   = (const float*)d_in[1];
  const float* v   = (const float*)d_in[2];
  const float* mbp = (const float*)d_in[3];
  const float* tbp = (const float*)d_in[4];
  float* o = (float*)d_out;

  dim3 grid(HEADS, NN);      // 1024 blocks, one (n,h) pair each
  TFA_triattn_kernel<<<grid, 128, 0, stream>>>(q, k, v, mbp, tbp, o);
}
